// MemoryConsolidation_51049981281159
// MI455X (gfx1250) — matrix operand order repaired and checked by reading, not compiled, not run on hardware
//
#include <hip/hip_runtime.h>
#include <hip/hip_bf16.h>
#include <math.h>

typedef __bf16 bf16;
typedef __attribute__((ext_vector_type(16))) __bf16       v16bf;
typedef __attribute__((ext_vector_type(8)))  float        v8f;
typedef __attribute__((ext_vector_type(4)))  unsigned int uint4v;

#define HIDDEN 2048
#define HALFD  1024
#define BATCH  4096
#define CAP    8192
#define LDS_STRIDE 40   // 32 K elems + 8 pad; 80B rows keep 16B alignment

// ---- CDNA5 async direct-to-LDS staging (ASYNCcnt path, CDNA5 ISA) ----
// LDS destination address comes from the VDST VGPR plus offset; addrspace(3) offset == low 32 bits
// of the generic pointer on gfx1250.
__device__ __forceinline__ unsigned lds_off(const void* p) {
  return (unsigned)(unsigned long long)p;
}
__device__ __forceinline__ void async_load_b128(unsigned dst, const void* gsrc) {
  asm volatile("global_load_async_to_lds_b128 %0, %1, off"
               :: "v"(dst), "v"((unsigned long long)gsrc) : "memory");
}
#define S_WAIT_ASYNCCNT(n) asm volatile("s_wait_asynccnt " #n ::: "memory")

// ------------------------------------------------------------------
// WMMA fragment loaders per CDNA5 ISA 16-bit layouts (CDNA5 ISA):
//  A 16x32 bf16: lanes 0-15 hold M=lane, K={0..7} in V0-3, {16..23} in
//  V4-7; lanes 16-31 hold K={8..15},{24..31} -> two b128 LDS loads.
//  B 32x16 bf16 (tile stored [n][k]): lanes 0-15 hold N=lane, K={0..7},{16..23};
//  lanes 16-31 hold K={8..15},{24..31} -> two b128 LDS loads.
// ------------------------------------------------------------------
union Frag32 { uint4v q[2]; v16bf v; };

__device__ __forceinline__ v16bf load_frag_a(const bf16* tile, int lane) {
  const int m = lane & 15;
  const int khalf = (lane >> 4) & 1;
  const bf16* p = tile + m * LDS_STRIDE + khalf * 8;
  Frag32 f;
  f.q[0] = *(const uint4v*)(p);
  f.q[1] = *(const uint4v*)(p + 16);
  return f.v;
}
__device__ __forceinline__ v16bf load_frag_b(const bf16* tile, int lane) {
  const int n = lane & 15;
  const int khalf = (lane >> 4) & 1;
  const bf16* p = tile + n * LDS_STRIDE + khalf * 8;
  Frag32 f;
  f.q[0] = *(const uint4v*)(p);
  f.q[1] = *(const uint4v*)(p + 16);
  return f.v;
}

enum { EPI_F32 = 0, EPI_BF16 = 1, EPI_RELU_BF16 = 2, EPI_SIG_ADD_F32 = 3 };

template<int EPI>
__device__ __forceinline__ void store_tile(v8f c, int grow0, int gcol,
                                           const float* __restrict__ bias,
                                           const float* __restrict__ resid,
                                           void* __restrict__ Cout, int ldc) {
  // C/D 16x16 f32 layout: VGPR v -> row (v + 8*(lane>=16)), col = lane&15
#pragma unroll
  for (int v = 0; v < 8; ++v) {
    const int grow = grow0 + v;
    const float val = c[v];
    if (EPI == EPI_F32) {
      ((float*)Cout)[(size_t)grow * ldc + gcol] = val;
    } else if (EPI == EPI_BF16) {
      ((bf16*)Cout)[(size_t)grow * ldc + gcol] = (bf16)val;
    } else if (EPI == EPI_RELU_BF16) {
      float r = val + bias[gcol];
      ((bf16*)Cout)[(size_t)grow * ldc + gcol] = (bf16)(r > 0.0f ? r : 0.0f);
    } else { // EPI_SIG_ADD_F32
      float g = 1.0f / (1.0f + __expf(-(val + bias[gcol])));
      ((float*)Cout)[(size_t)grow * ldc + gcol] = resid[(size_t)grow * ldc + gcol] + g;
    }
  }
}

// ------------------------------------------------------------------
// bf16 GEMM, f32 accumulate. C[M,N] = A[M,K] * op(B) (+ epilogue)
// TRANSB: B is [N,K] row-major (op(B)=B^T); else B is [K,N] row-major.
// Block tile 128x128, 8 waves (4x2), 32x64 per wave, K chunk 32.
// Double-buffered LDS fed by global_load_async_to_lds_b128.
// ------------------------------------------------------------------
template<int EPI, bool TRANSB>
__global__ __launch_bounds__(256) void gemm_bf16_wmma(
    const bf16* __restrict__ A, int lda,
    const bf16* __restrict__ B, int ldb,
    const float* __restrict__ bias,
    const float* __restrict__ resid,
    void* __restrict__ Cout, int ldc, int K)
{
  __shared__ bf16 Al[2][128 * LDS_STRIDE];
  __shared__ bf16 Bl[2][128 * LDS_STRIDE];

  const int tid  = threadIdx.x;
  const int lane = tid & 31;
  const int w    = tid >> 5;
  const int wm   = w >> 1;        // 0..3  (M direction, 32 rows each)
  const int wn   = w & 1;         // 0..1  (N direction, 64 cols each)
  const int blockM = blockIdx.y * 128;
  const int blockN = blockIdx.x * 128;

  // per-thread staging coordinates: 32B (16 bf16) per thread per operand
  const int arow = tid >> 1;           // 0..127
  const int acol = (tid & 1) * 16;     // 0 or 16

  auto stage = [&](int k0, int buf) {
    { // A tile 128x32 -> async direct to LDS (2 x b128)
      const bf16* ga = A + (size_t)(blockM + arow) * lda + k0 + acol;
      unsigned lo = lds_off(&Al[buf][arow * LDS_STRIDE + acol]);
      async_load_b128(lo, ga);
      async_load_b128(lo + 16, ga + 8);
    }
    if (TRANSB) { // B tile 128x32, already [n][k] in memory -> async (2 x b128)
      const bf16* gb = B + (size_t)(blockN + arow) * ldb + k0 + acol;
      unsigned lo = lds_off(&Bl[buf][arow * LDS_STRIDE + acol]);
      async_load_b128(lo, gb);
      async_load_b128(lo + 16, gb + 8);
    } else { // B tile 32x128 row-major -> transpose through VGPRs into [n][k]
      const int kk = tid >> 3;          // 0..31
      const int nn = (tid & 7) * 16;    // 0..112
      union { uint4v q[2]; bf16 h[16]; } bv;
      const bf16* gb = B + (size_t)(k0 + kk) * ldb + blockN + nn;
      bv.q[0] = *(const uint4v*)(gb);
      bv.q[1] = *(const uint4v*)(gb + 8);
#pragma unroll
      for (int j = 0; j < 16; ++j) Bl[buf][(nn + j) * LDS_STRIDE + kk] = bv.h[j];
    }
  };

  v8f acc[2][4];
#pragma unroll
  for (int mi = 0; mi < 2; ++mi)
#pragma unroll
    for (int ni = 0; ni < 4; ++ni) acc[mi][ni] = (v8f){};

  const int nk = K / 32;
  stage(0, 0);

  for (int i = 0; i < nk; ++i) {
    const int cur = i & 1;
    if (i + 1 < nk) {
      stage((i + 1) * 32, cur ^ 1);
      if (TRANSB) { S_WAIT_ASYNCCNT(4); } else { S_WAIT_ASYNCCNT(2); }
    } else {
      S_WAIT_ASYNCCNT(0);
    }
    __syncthreads();   // all waves' tile-i LDS writes visible

    const bf16* AlC = &Al[cur][0];
    const bf16* BlC = &Bl[cur][0];
    v16bf a0 = load_frag_a(AlC + (wm * 32)      * LDS_STRIDE, lane);
    v16bf a1 = load_frag_a(AlC + (wm * 32 + 16) * LDS_STRIDE, lane);
#pragma unroll
    for (int ni = 0; ni < 4; ++ni) {
      v16bf b = load_frag_b(BlC + (wn * 64 + ni * 16) * LDS_STRIDE, lane);
      acc[0][ni] = __builtin_amdgcn_wmma_f32_16x16x32_bf16(false, a0, false, b, (short)0, acc[0][ni], false, false);
      acc[1][ni] = __builtin_amdgcn_wmma_f32_16x16x32_bf16(false, a1, false, b, (short)0, acc[1][ni], false, false);
    }
    __syncthreads();   // protect buffer before next round overwrites it
  }

  const int colInTile = lane & 15;
  const int rowHalf   = (lane >> 4) * 8;
#pragma unroll
  for (int mi = 0; mi < 2; ++mi)
#pragma unroll
    for (int ni = 0; ni < 4; ++ni) {
      const int grow0 = blockM + wm * 32 + mi * 16 + rowHalf;
      const int gcol0 = blockN + wn * 64 + ni * 16 + colInTile;
      store_tile<EPI>(acc[mi][ni], grow0, gcol0, bias, resid, Cout, ldc);
    }
}

// ------------------------------------------------------------------
// Row softmax over S[4096,4096] with the CAPACITY zero-row correction:
//   m = max(max_j s_j, 0);  denom = sum_j exp(s_j - m) + 4096*exp(-m)
// Row staged in LDS so P (bf16) is written in place over the same 16KB
// row slot (P row stride = 8192 bf16 elements).
// ------------------------------------------------------------------
__global__ __launch_bounds__(256) void softmax_inplace(float* __restrict__ SP) {
  __shared__ float row[BATCH];
  __shared__ float redmax[8];
  __shared__ float redsum[8];
  const int tid = threadIdx.x, lane = tid & 31, w = tid >> 5;
  float* Srow = SP + (size_t)blockIdx.x * BATCH;
  bf16*  Prow = (bf16*)SP + (size_t)blockIdx.x * (2 * BATCH);

  float lmax = 0.0f;  // zero memory rows have sim == 0
  for (int c = tid; c < BATCH; c += 256) {
    float v = Srow[c];
    row[c] = v;
    lmax = fmaxf(lmax, v);
  }
#pragma unroll
  for (int off = 16; off; off >>= 1) lmax = fmaxf(lmax, __shfl_xor(lmax, off, 32));
  if (lane == 0) redmax[w] = lmax;
  __syncthreads();
  float m = redmax[0];
#pragma unroll
  for (int i = 1; i < 8; ++i) m = fmaxf(m, redmax[i]);

  float lsum = 0.0f;
  for (int c = tid; c < BATCH; c += 256) lsum += __expf(row[c] - m);
#pragma unroll
  for (int off = 16; off; off >>= 1) lsum += __shfl_xor(lsum, off, 32);
  if (lane == 0) redsum[w] = lsum;
  __syncthreads();
  float denom = (float)(CAP - BATCH) * __expf(-m);  // 4096 zero rows
#pragma unroll
  for (int i = 0; i < 8; ++i) denom += redsum[i];
  const float inv = 1.0f / denom;

  for (int c = tid; c < BATCH; c += 256)
    Prow[c] = (bf16)(__expf(row[c] - m) * inv);
}

__global__ __launch_bounds__(256) void cvt_f32_to_bf16(const float* __restrict__ in,
                                                       bf16* __restrict__ out, int n) {
  int i = blockIdx.x * blockDim.x + threadIdx.x;
  const int stride = gridDim.x * blockDim.x;
  for (; i < n; i += stride) out[i] = (bf16)in[i];
}

// ------------------------------------------------------------------
// Pipeline:
//  1) x,W1,W2 -> bf16
//  2) S  = xb*xb^T                       (f32 out, 67MB, L2-resident)
//  3) P  = softmax(S) in place           (bf16, row stride 8192)
//  4) ret= P*xb                          (bf16 out)
//  5) h  = relu(ret*W1^T + b1)           (bf16 out)
//  6) out= x + sigmoid(h*W2^T + b2)      (f32 out)
// Workspace: ~117.5 MB.
// ------------------------------------------------------------------
extern "C" void kernel_launch(void* const* d_in, const int* in_sizes, int n_in,
                              void* d_out, int out_size, void* d_ws, size_t ws_size,
                              hipStream_t stream) {
  const float* x  = (const float*)d_in[0];
  const float* W1 = (const float*)d_in[1];
  const float* b1 = (const float*)d_in[2];
  const float* W2 = (const float*)d_in[3];
  const float* b2 = (const float*)d_in[4];
  float* out = (float*)d_out;
  (void)in_sizes; (void)n_in; (void)out_size; (void)ws_size;

  char* p = (char*)d_ws;
  bf16*  xb   = (bf16*)p;  p += (size_t)BATCH * HIDDEN * sizeof(bf16);
  float* SP   = (float*)p; p += (size_t)BATCH * BATCH  * sizeof(float);
  bf16*  retb = (bf16*)p;  p += (size_t)BATCH * HIDDEN * sizeof(bf16);
  bf16*  hb   = (bf16*)p;  p += (size_t)BATCH * HALFD  * sizeof(bf16);
  bf16*  w1b  = (bf16*)p;  p += (size_t)HALFD * HIDDEN * sizeof(bf16);
  bf16*  w2b  = (bf16*)p;  p += (size_t)HIDDEN * HALFD * sizeof(bf16);

  cvt_f32_to_bf16<<<2048, 256, 0, stream>>>(x,  xb,  BATCH * HIDDEN);
  cvt_f32_to_bf16<<<1024, 256, 0, stream>>>(W1, w1b, HALFD * HIDDEN);
  cvt_f32_to_bf16<<<1024, 256, 0, stream>>>(W2, w2b, HIDDEN * HALFD);

  // S = xb * xb^T   (B operand = xb rows as columns -> TRANSB)
  gemm_bf16_wmma<EPI_F32, true><<<dim3(BATCH / 128, BATCH / 128), 256, 0, stream>>>(
      xb, HIDDEN, xb, HIDDEN, nullptr, nullptr, SP, BATCH, HIDDEN);

  softmax_inplace<<<BATCH, 256, 0, stream>>>(SP);

  // retrieved = P * xb   (P lda = 8192 bf16 elems due to in-place slots)
  gemm_bf16_wmma<EPI_BF16, false><<<dim3(HIDDEN / 128, BATCH / 128), 256, 0, stream>>>(
      (const bf16*)SP, 2 * BATCH, xb, HIDDEN, nullptr, nullptr, retb, HIDDEN, BATCH);

  // h = relu(ret * W1^T + b1)
  gemm_bf16_wmma<EPI_RELU_BF16, true><<<dim3(HALFD / 128, BATCH / 128), 256, 0, stream>>>(
      retb, HIDDEN, w1b, HIDDEN, b1, nullptr, hb, HALFD, HIDDEN);

  // out = x + sigmoid(h * W2^T + b2)
  gemm_bf16_wmma<EPI_SIG_ADD_F32, true><<<dim3(HIDDEN / 128, BATCH / 128), 256, 0, stream>>>(
      hb, HALFD, w2b, HALFD, b2, x, out, HIDDEN, HALFD);
}
